// WarlightPolicyNetSAGE_87995289960625
// MI455X (gfx1250) — matrix operand order repaired and checked by reading, not compiled, not run on hardware
//
#include <hip/hip_runtime.h>

#define NN 50000
#define EE 1600000
#define BBATCH 16
#define AACT 4096
#define DD 64
#define KK 50
#define BA (BBATCH*AACT)

typedef __attribute__((ext_vector_type(16))) _Float16 v16h;
typedef __attribute__((ext_vector_type(8)))  float    v8f;
typedef unsigned int u32x4 __attribute__((ext_vector_type(4)));
typedef int          i32x4 __attribute__((ext_vector_type(4)));
typedef int          i32x8 __attribute__((ext_vector_type(8)));

#if defined(__has_builtin)
#if __has_builtin(__builtin_amdgcn_tensor_load_to_lds)
#define USE_TDM 1
#endif
#endif

__device__ __forceinline__ float clipf(float x, float lo, float hi) {
  return fminf(fmaxf(x, lo), hi);
}

// ---- A fragment (16x32 f16): lane m=lane&15 holds row m; hi=lane>>4 selects
// K chunks [8*hi, 8*hi+8) and [16+8*hi, 16+8*hi+8) per ISA 7.12.2 layout.
__device__ __forceinline__ v16h load_a_row(const float* __restrict__ rowp, int kbase,
                                           int hi, float scale) {
  v16h a;
  const float* p0 = rowp + kbase + 8*hi;
  const float* p1 = rowp + kbase + 16 + 8*hi;
#pragma unroll
  for (int j = 0; j < 8; ++j) {
    a[j]   = (_Float16)(p0[j] * scale);
    a[j+8] = (_Float16)(p1[j] * scale);
  }
  return a;
}

// A fragment for the concatenated [ne[src] | ne[tgt]] row (128 wide), clip +-3.
__device__ __forceinline__ v16h load_a_pair_clip(const float* __restrict__ r0,
                                                 const float* __restrict__ r1,
                                                 int kbase, int hi) {
  v16h a;
  int b0 = kbase + 8*hi, b1 = kbase + 16 + 8*hi;
  const float* p0 = (b0 < DD) ? (r0 + b0) : (r1 + (b0 - DD));
  const float* p1 = (b1 < DD) ? (r0 + b1) : (r1 + (b1 - DD));
#pragma unroll
  for (int j = 0; j < 8; ++j) {
    a[j]   = (_Float16)clipf(p0[j], -3.f, 3.f);
    a[j+8] = (_Float16)clipf(p1[j], -3.f, 3.f);
  }
  return a;
}

// ---- B fragment from LDS-staged transposed f16 weights wT[col*KD + k]:
// lane holds col, elements e<8 -> K = kbase + 8*hi + e, e>=8 -> K = kbase + 16 + 8*hi + (e-8)
// (two 8-half chunks, 16B aligned) => two ds_load_b128 per fragment.
__device__ __forceinline__ v16h load_b_lds(const _Float16* __restrict__ wT, int KD,
                                           int kbase, int col, int hi) {
  v16h b;
  const _Float16* q0 = wT + col*KD + kbase + 8*hi;
  const _Float16* q1 = wT + col*KD + kbase + 16 + 8*hi;
#pragma unroll
  for (int j = 0; j < 8; ++j) { b[j] = q0[j]; b[j+8] = q1[j]; }
  return b;
}

// ---- B fragment straight from global f32 (small heads only).
__device__ __forceinline__ v16h load_b_frag(const float* __restrict__ W, int ldw,
                                            int kbase, int col, int hi, int ncols) {
  v16h b;
#pragma unroll
  for (int e = 0; e < 16; ++e) {
    int k = kbase + ((e < 8) ? (8*hi + e) : (16 + 8*hi + (e - 8)));
    float w = (col < ncols) ? W[k*ldw + col] : 0.0f;
    b[e] = (_Float16)w;
  }
  return b;
}

#ifdef USE_TDM
// TDM: DMA a 2-D f32 tensor (dim1 rows x dim0 cols, contiguous) into LDS.
// D# per CDNA5 ISA 08_async_tensor 8.3/8.4 (group0: count|lds|global|type,
// group1: data_size, tensor dims, tile dims, stride).
__device__ __forceinline__ void tdm_load_2d_f32(unsigned int ldsOff, const void* gptr,
                                                unsigned int dim0, unsigned int dim1) {
  unsigned long long ga = (unsigned long long)gptr;
  u32x4 g0;
  g0[0] = 1u;                                              // count=1 (valid user D#)
  g0[1] = ldsOff;                                          // lds_addr (bytes)
  g0[2] = (unsigned int)(ga & 0xffffffffu);                // global_addr[31:0]
  g0[3] = (unsigned int)((ga >> 32) & 0x1ffffffu) | (2u << 30); // addr[56:32]|type=2
  i32x8 g1;
  g1[0] = (int)(2u << 16);                                 // data_size=2 (4B)
  g1[1] = (int)(dim0 << 16);                               // tensor_dim0[15:0] @63:48
  g1[2] = (int)((dim0 >> 16) | (dim1 << 16));              // dim0 hi | tensor_dim1 lo
  g1[3] = (int)((dim1 >> 16) | (dim0 << 16));              // dim1 hi | tile_dim0
  g1[4] = (int)dim1;                                       // tile_dim1 (tile_dim2=0)
  g1[5] = (int)dim0;                                       // tensor_dim0_stride lo32
  g1[6] = 0;
  g1[7] = 0;
  i32x4 gz; gz[0]=0; gz[1]=0; gz[2]=0; gz[3]=0;
#if __clang_major__ >= 23
  i32x8 gz8; gz8[0]=0; gz8[1]=0; gz8[2]=0; gz8[3]=0; gz8[4]=0; gz8[5]=0; gz8[6]=0; gz8[7]=0;
  __builtin_amdgcn_tensor_load_to_lds(g0, g1, gz, gz, gz8, 0);
#else
  __builtin_amdgcn_tensor_load_to_lds(g0, g1, gz, gz, 0);
#endif
}
#endif

__global__ void zero_f32(float* __restrict__ p, long n) {
  long i = blockIdx.x * (long)blockDim.x + threadIdx.x;
  long stride = (long)gridDim.x * blockDim.x;
  for (; i < n; i += stride) p[i] = 0.0f;
}

// Scatter-add h[src] into acc[dst] (+count). One thread per (edge, 4-dim chunk).
// h fits L2 (12.8MB of 192MB) so these atomics are L2-resident.
template<int DIN>
__global__ void edge_scatter(const float* __restrict__ h, const int* __restrict__ ei,
                             float* __restrict__ acc, float* __restrict__ cnt) {
  const int CH = DIN/4;
  long idx = blockIdx.x * (long)blockDim.x + threadIdx.x;
  if (idx >= (long)EE * CH) return;
  int e = (int)(idx / CH);
  int c = (int)(idx & (CH-1));
  int s = ei[e], d = ei[EE + e];
  const float4 v = *(const float4*)(h + (long)s*DIN + c*4);
  float* ap = acc + (long)d*DIN + c*4;
  atomicAdd(ap+0, v.x); atomicAdd(ap+1, v.y);
  atomicAdd(ap+2, v.z); atomicAdd(ap+3, v.w);
  if (c == 0) atomicAdd(&cnt[d], 1.0f);
}

// Fused SAGE layer: out = relu(bn(l2norm(mean@wl + bl + h@wr))) [+clip 5 if last].
// Weights TDM-staged to LDS once per block, converted to transposed f16; each
// wave then does one 16-row tile with WMMA (B frags = 2x ds_load_b128).
template<int DIN>
__global__ void sage_gemm(const float* __restrict__ h, const float* __restrict__ acc,
                          const float* __restrict__ cnt,
                          const float* __restrict__ wl, const float* __restrict__ bl,
                          const float* __restrict__ wr,
                          const float* __restrict__ gg, const float* __restrict__ bbn,
                          const float* __restrict__ mm, const float* __restrict__ vv,
                          float* __restrict__ out, int doClip) {
  __shared__ float rawW[2][64*DD];                                   // f32 [k][col]
  __shared__ _Float16 wT[2][DD*DD] __attribute__((aligned(32)));     // f16 [col][k]

  // ---- stage weights (all threads reach both barriers; no early return yet)
#ifdef USE_TDM
  if ((threadIdx.x >> 5) == 0) {
    tdm_load_2d_f32((unsigned int)(unsigned long long)&rawW[0][0], wl, DD, DIN);
    tdm_load_2d_f32((unsigned int)(unsigned long long)&rawW[1][0], wr, DD, DIN);
    __builtin_amdgcn_s_wait_tensorcnt(0);
  }
  __syncthreads();
  for (int idx = threadIdx.x; idx < DIN*DD; idx += blockDim.x) {
    int k = idx >> 6, col = idx & 63;
    wT[0][col*DIN + k] = (_Float16)rawW[0][idx];
    wT[1][col*DIN + k] = (_Float16)rawW[1][idx];
  }
#else
  for (int idx = threadIdx.x; idx < DIN*DD; idx += blockDim.x) {
    int k = idx >> 6, col = idx & 63;
    wT[0][col*DIN + k] = (_Float16)wl[idx];
    wT[1][col*DIN + k] = (_Float16)wr[idx];
  }
#endif
  __syncthreads();

  int wave = (int)((blockIdx.x * (long)blockDim.x + threadIdx.x) >> 5);
  if (wave >= NN/16) return;
  int lane = threadIdx.x & 31;
  int m = lane & 15, hi = lane >> 4;
  int row = wave*16 + m;
  float invc = 1.0f / fmaxf(cnt[row], 1.0f);
  const float* accRow = acc + (long)row*DIN;
  const float* hRow   = h   + (long)row*DIN;

  v8f c[4];
#pragma unroll
  for (int t = 0; t < 4; ++t)
#pragma unroll
    for (int r = 0; r < 8; ++r) c[t][r] = 0.0f;

#pragma unroll
  for (int ks = 0; ks < DIN/32; ++ks) {             // mean @ wl
    v16h a = load_a_row(accRow, ks*32, hi, invc);
#pragma unroll
    for (int t = 0; t < 4; ++t) {
      v16h b = load_b_lds(wT[0], DIN, ks*32, t*16 + m, hi);
      c[t] = __builtin_amdgcn_wmma_f32_16x16x32_f16(false, a, false, b, (short)0, c[t], false, false);
    }
  }
#pragma unroll
  for (int ks = 0; ks < DIN/32; ++ks) {             // h @ wr
    v16h a = load_a_row(hRow, ks*32, hi, 1.0f);
#pragma unroll
    for (int t = 0; t < 4; ++t) {
      v16h b = load_b_lds(wT[1], DIN, ks*32, t*16 + m, hi);
      c[t] = __builtin_amdgcn_wmma_f32_16x16x32_f16(false, a, false, b, (short)0, c[t], false, false);
    }
  }
  // + bias
  float blv[4];
#pragma unroll
  for (int t = 0; t < 4; ++t) blv[t] = bl[t*16 + m];
#pragma unroll
  for (int t = 0; t < 4; ++t)
#pragma unroll
    for (int r = 0; r < 8; ++r) c[t][r] += blv[t];
  // L2 row norm: reduce across the 16 lanes of each half (shfl_xor 1|2|4|8 stays in-half)
  float invn[8];
#pragma unroll
  for (int r = 0; r < 8; ++r) {
    float s = 0.f;
#pragma unroll
    for (int t = 0; t < 4; ++t) s += c[t][r]*c[t][r];
    s += __shfl_xor(s, 1, 32);
    s += __shfl_xor(s, 2, 32);
    s += __shfl_xor(s, 4, 32);
    s += __shfl_xor(s, 8, 32);
    invn[r] = 1.0f / fmaxf(sqrtf(s), 1e-12f);
  }
  float sc[4], sh[4];
#pragma unroll
  for (int t = 0; t < 4; ++t) {
    int col = t*16 + m;
    float s1 = gg[col] * rsqrtf(vv[col] + 1e-5f);
    sc[t] = s1;
    sh[t] = bbn[col] - mm[col]*s1;
  }
#pragma unroll
  for (int t = 0; t < 4; ++t) {
    int col = t*16 + m;
#pragma unroll
    for (int r = 0; r < 8; ++r) {
      float y = c[t][r] * invn[r];
      y = fmaf(y, sc[t], sh[t]);
      y = fmaxf(y, 0.0f);
      if (doClip) y = fminf(y, 5.0f);
      out[(long)(wave*16 + r + 8*hi)*DD + col] = y;
    }
  }
}

// Placement head: logits = clip(relu(bn(ne@w1 + b1)) @ w2 + b2, -15, 15)
__global__ void placement_head(const float* __restrict__ ne,
                               const float* __restrict__ w1, const float* __restrict__ b1,
                               const float* __restrict__ gg, const float* __restrict__ bbn,
                               const float* __restrict__ mm, const float* __restrict__ vv,
                               const float* __restrict__ w2, const float* __restrict__ b2,
                               float* __restrict__ out) {
  int wave = (int)((blockIdx.x * (long)blockDim.x + threadIdx.x) >> 5);
  if (wave >= NN/16) return;
  int lane = threadIdx.x & 31;
  int m = lane & 15, hi = lane >> 4;
  int row = wave*16 + m;
  const float* neRow = ne + (long)row*DD;
  v8f c[2];
#pragma unroll
  for (int t = 0; t < 2; ++t)
#pragma unroll
    for (int r = 0; r < 8; ++r) c[t][r] = 0.0f;
#pragma unroll
  for (int ks = 0; ks < 2; ++ks) {
    v16h a = load_a_row(neRow, ks*32, hi, 1.0f);
#pragma unroll
    for (int t = 0; t < 2; ++t) {
      v16h b = load_b_frag(w1, 32, ks*32, t*16 + m, hi, 32);
      c[t] = __builtin_amdgcn_wmma_f32_16x16x32_f16(false, a, false, b, (short)0, c[t], false, false);
    }
  }
  float b1v[2], sc[2], sh[2], w2v[2];
#pragma unroll
  for (int t = 0; t < 2; ++t) {
    int col = t*16 + m;
    b1v[t] = b1[col];
    float s1 = gg[col] * rsqrtf(vv[col] + 1e-5f);
    sc[t] = s1;
    sh[t] = bbn[col] - mm[col]*s1;
    w2v[t] = w2[col];
  }
#pragma unroll
  for (int r = 0; r < 8; ++r) {
    float dot = 0.f;
#pragma unroll
    for (int t = 0; t < 2; ++t) {
      float p = fmaxf(fmaf(c[t][r] + b1v[t], sc[t], sh[t]), 0.0f);
      dot += p * w2v[t];
    }
    dot += __shfl_xor(dot, 1, 32);
    dot += __shfl_xor(dot, 2, 32);
    dot += __shfl_xor(dot, 4, 32);
    dot += __shfl_xor(dot, 8, 32);
    if (m == 0) out[wave*16 + r + 8*hi] = clipf(dot + b2[0], -15.f, 15.f);
  }
}

// Fully fused action-edge head. All three weight matrices are staged once per
// block into LDS as transposed f16 (dst-order: conflict-free LDS writes,
// cached global reads), so every B fragment is 2x ds_load_b128.
__global__ void edge_head(const float* __restrict__ ne, const int* __restrict__ ae,
                          const unsigned char* __restrict__ mask,
                          const float* __restrict__ ew1, const float* __restrict__ eb1,
                          const float* __restrict__ eg,  const float* __restrict__ ebb,
                          const float* __restrict__ em,  const float* __restrict__ ev,
                          const float* __restrict__ ew2, const float* __restrict__ eb2,
                          const float* __restrict__ aw1, const float* __restrict__ ab1,
                          const float* __restrict__ ag,  const float* __restrict__ abb,
                          const float* __restrict__ am,  const float* __restrict__ av,
                          const float* __restrict__ aw2, const float* __restrict__ ab2,
                          float* __restrict__ attack, float* __restrict__ army) {
  __shared__ _Float16 ewT[64*128]  __attribute__((aligned(32))); // es_w1^T  16KB
  __shared__ _Float16 awT[64*128]  __attribute__((aligned(32))); // as_w1^T  16KB
  __shared__ _Float16 aw2T[64*64]  __attribute__((aligned(32))); // as_w2^T (padded) 8KB
  __shared__ _Float16 ldsbuf[8][16*DD];                          // army C->A relayout 16KB

  for (int idx = threadIdx.x; idx < 64*128; idx += blockDim.x) {
    int col = idx >> 7, k = idx & 127;
    ewT[idx] = (_Float16)ew1[k*DD + col];
    awT[idx] = (_Float16)aw1[k*DD + col];
  }
  for (int idx = threadIdx.x; idx < 64*64; idx += blockDim.x) {
    int col = idx >> 6, k = idx & 63;
    aw2T[idx] = (_Float16)((col < KK) ? aw2[k*KK + col] : 0.0f);
  }
  __syncthreads();

  int wave = (int)((blockIdx.x * (long)blockDim.x + threadIdx.x) >> 5);
  if (wave >= BA/16) return;
  int wv = threadIdx.x >> 5;
  int lane = threadIdx.x & 31;
  int m = lane & 15, hi = lane >> 4;
  int i = wave*16 + m;
  int sN = min(max(ae[2*i],   0), NN-1);
  int tN = min(max(ae[2*i+1], 0), NN-1);
  const float* sRow = ne + (long)sN*DD;
  const float* tRow = ne + (long)tN*DD;

  v8f ce[4], ca[4];
#pragma unroll
  for (int t = 0; t < 4; ++t)
#pragma unroll
    for (int r = 0; r < 8; ++r) { ce[t][r] = 0.0f; ca[t][r] = 0.0f; }

#pragma unroll
  for (int ks = 0; ks < 4; ++ks) {                    // K = 128, shared A frag
    v16h a = load_a_pair_clip(sRow, tRow, ks*32, hi);
#pragma unroll
    for (int t = 0; t < 4; ++t) {
      v16h b = load_b_lds(ewT, 128, ks*32, t*16 + m, hi);
      ce[t] = __builtin_amdgcn_wmma_f32_16x16x32_f16(false, a, false, b, (short)0, ce[t], false, false);
    }
#pragma unroll
    for (int t = 0; t < 4; ++t) {
      v16h b = load_b_lds(awT, 128, ks*32, t*16 + m, hi);
      ca[t] = __builtin_amdgcn_wmma_f32_16x16x32_f16(false, a, false, b, (short)0, ca[t], false, false);
    }
  }

  // ---- attack head: dot(relu(bn(ce+b1)), w2) per row
  float b1E[4], scE[4], shE[4], w2E[4];
#pragma unroll
  for (int t = 0; t < 4; ++t) {
    int col = t*16 + m;
    b1E[t] = eb1[col];
    float s1 = eg[col] * rsqrtf(ev[col] + 1e-5f);
    scE[t] = s1;
    shE[t] = ebb[col] - em[col]*s1;
    w2E[t] = ew2[col];
  }
#pragma unroll
  for (int r = 0; r < 8; ++r) {
    float dot = 0.f;
#pragma unroll
    for (int t = 0; t < 4; ++t) {
      float e2 = fmaxf(fmaf(ce[t][r] + b1E[t], scE[t], shE[t]), 0.0f);
      dot += e2 * w2E[t];
    }
    dot += __shfl_xor(dot, 1, 32);
    dot += __shfl_xor(dot, 2, 32);
    dot += __shfl_xor(dot, 4, 32);
    dot += __shfl_xor(dot, 8, 32);
    if (m == 0) {
      int rr = wave*16 + r + 8*hi;
      float al = clipf(dot + eb2[0], -15.f, 15.f);
      attack[rr] = mask[rr] ? al : -1e9f;
    }
  }

  // ---- army head: a = relu(bn(ca+b1)) -> LDS (f16, row-major) -> A frags -> @aw2
  float b1A[4], scA[4], shA[4];
#pragma unroll
  for (int t = 0; t < 4; ++t) {
    int col = t*16 + m;
    b1A[t] = ab1[col];
    float s1 = ag[col] * rsqrtf(av[col] + 1e-5f);
    scA[t] = s1;
    shA[t] = abb[col] - am[col]*s1;
  }
  _Float16* L = ldsbuf[wv];
#pragma unroll
  for (int t = 0; t < 4; ++t)
#pragma unroll
    for (int r = 0; r < 8; ++r) {
      float y = fmaxf(fmaf(ca[t][r] + b1A[t], scA[t], shA[t]), 0.0f);
      L[(r + 8*hi)*DD + t*16 + m] = (_Float16)y;
    }
  asm volatile("s_wait_dscnt 0" ::: "memory");       // same-wave LDS RAW ordering

  v8f co[4];
#pragma unroll
  for (int t = 0; t < 4; ++t)
#pragma unroll
    for (int r = 0; r < 8; ++r) co[t][r] = 0.0f;
#pragma unroll
  for (int ks = 0; ks < 2; ++ks) {                    // K = 64
    v16h a;
#pragma unroll
    for (int j = 0; j < 8; ++j) {
      a[j]   = L[m*DD + ks*32 + 8*hi + j];
      a[j+8] = L[m*DD + ks*32 + 16 + 8*hi + j];
    }
#pragma unroll
    for (int t = 0; t < 4; ++t) {
      v16h b = load_b_lds(aw2T, DD, ks*32, t*16 + m, hi);
      co[t] = __builtin_amdgcn_wmma_f32_16x16x32_f16(false, a, false, b, (short)0, co[t], false, false);
    }
  }
#pragma unroll
  for (int t = 0; t < 4; ++t) {
    int col = t*16 + m;
    float bias2 = (col < KK) ? ab2[col] : 0.0f;
#pragma unroll
    for (int r = 0; r < 8; ++r) {
      if (col < KK) {
        int rr = wave*16 + r + 8*hi;
        float y = clipf(co[t][r] + bias2, -15.f, 15.f);
        army[(long)rr*KK + col] = mask[rr] ? y : -1e9f;
      }
    }
  }
}

extern "C" void kernel_launch(void* const* d_in, const int* in_sizes, int n_in,
                              void* d_out, int out_size, void* d_ws, size_t ws_size,
                              hipStream_t stream) {
  const float* x = (const float*)d_in[0];
  const int* ei  = (const int*)d_in[1];
  const int* ae  = (const int*)d_in[2];
  const unsigned char* mask = (const unsigned char*)d_in[4];
  auto P = [&](int i){ return (const float*)d_in[i]; };

  float* ws  = (float*)d_ws;
  float* acc = ws;                          // N*64 f32
  float* cnt = ws + (long)NN*DD;            // N f32
  float* hA  = cnt + NN;                    // N*64 f32
  float* hB  = hA + (long)NN*DD;            // N*64 f32
  float* outp = (float*)d_out;

  const int TB = 256;
  const long zn = (long)NN*DD + NN;         // acc + cnt contiguous
  const int gemmBlocks = (NN/16 + 7)/8;     // 8 waves per block, wave per 16 rows

  // ---- SAGE layer 1 (din = 32, input x)
  zero_f32<<<512, TB, 0, stream>>>(acc, zn);
  edge_scatter<32><<<(int)(((long)EE*8 + TB-1)/TB), TB, 0, stream>>>(x, ei, acc, cnt);
  sage_gemm<32><<<gemmBlocks, TB, 0, stream>>>(x, acc, cnt,
      P(5), P(6), P(7), P(8), P(9), P(10), P(11), hA, 0);

  // ---- SAGE layer 2 (din = 64)
  zero_f32<<<512, TB, 0, stream>>>(acc, zn);
  edge_scatter<64><<<(int)(((long)EE*16 + TB-1)/TB), TB, 0, stream>>>(hA, ei, acc, cnt);
  sage_gemm<64><<<gemmBlocks, TB, 0, stream>>>(hA, acc, cnt,
      P(12), P(13), P(14), P(15), P(16), P(17), P(18), hB, 0);

  // ---- SAGE layer 3 (din = 64) + clip(-5,5) -> ne in hA
  zero_f32<<<512, TB, 0, stream>>>(acc, zn);
  edge_scatter<64><<<(int)(((long)EE*16 + TB-1)/TB), TB, 0, stream>>>(hB, ei, acc, cnt);
  sage_gemm<64><<<gemmBlocks, TB, 0, stream>>>(hB, acc, cnt,
      P(19), P(20), P(21), P(22), P(23), P(24), P(25), hA, 1);

  // ---- heads
  placement_head<<<gemmBlocks, TB, 0, stream>>>(hA,
      P(26), P(27), P(28), P(29), P(30), P(31), P(32), P(33), outp);

  edge_head<<<BA/16/8, TB, 0, stream>>>(hA, ae, mask,
      P(34), P(35), P(36), P(37), P(38), P(39), P(40), P(41),
      P(42), P(43), P(44), P(45), P(46), P(47), P(48), P(49),
      outp + NN, outp + NN + BA);
}
